// SwinBlock_63385127354491
// MI455X (gfx1250) — compile-verified, matrix operand order checked by reading, not run on hardware
//
#include <hip/hip_runtime.h>

#define DIM   180
#define NHEAD 6
#define HD    30
#define SS    8
#define NN    256
#define NWIN  64
#define HID   360
#define QKVN  540

typedef __attribute__((ext_vector_type(16))) __bf16 v16bf;
typedef __attribute__((ext_vector_type(8)))  float  v8f;

// Element index inside a 512-element 16-bit WMMA fragment (ISA 7.12.2).
// For A-fragments call as frag_idx(row m, k); for B-fragments as frag_idx(col n, k).
__device__ __forceinline__ int frag_idx(int mn, int k) {
  return (mn + (((k >> 3) & 1) << 4)) * 16 + ((k >> 4) << 3) + (k & 7);
}

__device__ __forceinline__ v8f wmma_bf16(v16bf a, v16bf b, v8f c) {
  return __builtin_amdgcn_wmma_f32_16x16x32_bf16(false, a, false, b, (short)0, c,
                                                 false, false);
}

__device__ __forceinline__ float gelu_f(float x) {
  return 0.5f * x * (1.0f + erff(x * 0.70710678118654752f));
}

// ---------------------------------------------------------------------------
// P0: pack an fp32 [K][N] weight matrix into bf16 WMMA B-fragments laid out
//     [ntile][kstep][512], pads zero-filled. One thread per fragment element.
// ---------------------------------------------------------------------------
__global__ void __launch_bounds__(256) k_pack_w(
    const float* __restrict__ src, __bf16* __restrict__ dst, int K, int N,
    int ntiles, int ksteps) {
  long long idx = (long long)blockIdx.x * 256 + threadIdx.x;
  long long total = (long long)ntiles * ksteps * 512;
  if (idx >= total) return;
  int e = (int)(idx & 511);
  long long f = idx >> 9;
  int kstep = (int)(f % ksteps);
  int ntile = (int)(f / ksteps);
  int lane = e >> 4, i = e & 15;
  int nl = lane & 15, half = lane >> 4;
  int k = kstep * 32 + ((i >> 3) << 4) + (half << 3) + (i & 7);
  int n = ntile * 16 + nl;
  dst[idx] = (k < K && n < N) ? (__bf16)src[(long long)k * N + n] : (__bf16)0.0f;
}

// P1: flatten rpi->rpb into bias[head][q][k] (fp32, 1.5MB, L2-resident).
__global__ void __launch_bounds__(256) k_bias_tab(
    const int* __restrict__ rpi, const float* __restrict__ rpb,
    float* __restrict__ biasT) {
  int idx = blockIdx.x * 256 + threadIdx.x;
  if (idx >= NHEAD * NN * NN) return;
  int head = idx >> 16, qk = idx & 65535;
  biasT[idx] = rpb[rpi[qk] * NHEAD + head];
}

// ---------------------------------------------------------------------------
// K1: LayerNorm1 + QKV GEMM (180(pad192) x 540), scatter q/k/v into
//     WMMA-fragment-ordered per-(window,head) buffers with the -8 shift fused.
// ---------------------------------------------------------------------------
__global__ void __launch_bounds__(256) k_ln_qkv(
    const float* __restrict__ x, const float* __restrict__ nw,
    const float* __restrict__ nb, const __bf16* __restrict__ wq,
    const float* __restrict__ bqkv, __bf16* __restrict__ qb,
    __bf16* __restrict__ kb, __bf16* __restrict__ vb) {
  __shared__ float xs[16][DIM];
  __shared__ __align__(32) __bf16 af[6][512];
  const int tid = threadIdx.x;
  const long long g0 = (long long)blockIdx.x * 16;

  for (int idx = tid; idx < 16 * DIM; idx += 256) {
    int r = idx / DIM, c = idx % DIM;
    xs[r][c] = x[(g0 + r) * DIM + c];
  }
  __syncthreads();
  if (tid < 16) {
    int r = tid;
    float m = 0.f;
    for (int c = 0; c < DIM; ++c) m += xs[r][c];
    m *= (1.0f / DIM);
    float v = 0.f;
    for (int c = 0; c < DIM; ++c) { float d = xs[r][c] - m; v += d * d; }
    float rs = rsqrtf(v * (1.0f / DIM) + 1e-5f);
    for (int k = 0; k < 192; ++k) {
      float val = (k < DIM) ? (xs[r][k] - m) * rs * nw[k] + nb[k] : 0.f;
      af[k >> 5][frag_idx(r, k & 31)] = (__bf16)val;
    }
  }
  __syncthreads();

  const int lane = tid & 31, wave = tid >> 5;
  v16bf A[6];
#pragma unroll
  for (int kk = 0; kk < 6; ++kk) A[kk] = *(const v16bf*)&af[kk][lane * 16];
  const int nl = lane & 15, half = lane >> 4;

  for (int nt = wave; nt < 34; nt += 8) {
    int n = nt * 16 + nl;
    const __bf16* bp = wq + (long long)nt * 6 * 512 + lane * 16;
    // Preload all k-step B fragments: one load clause, one wait, then 6 WMMAs.
    v16bf Bf[6];
#pragma unroll
    for (int kk = 0; kk < 6; ++kk) Bf[kk] = *(const v16bf*)(bp + (kk << 9));
    v8f acc = {};
#pragma unroll
    for (int kk = 0; kk < 6; ++kk) acc = wmma_bf16(A[kk], Bf[kk], acc);
    if (n < QKVN) {
      float bias = bqkv[n];
      int sel = n / DIM, wi = n % DIM, head = wi / HD, d = wi % HD;
#pragma unroll
      for (int j = 0; j < 8; ++j) {
        int m = j + (half << 3);
        long long g = g0 + m;
        int b = (int)(g >> 14), rem = (int)(g & 16383);
        int h = rem >> 7, w = rem & 127;
        int hr = (h - SS) & 127, wr = (w - SS) & 127;
        int win = b * NWIN + ((hr >> 4) << 3) + (wr >> 4);
        int intra = ((hr & 15) << 4) + (wr & 15);
        float c = acc[j] + bias;
        if (sel == 0) {
          c *= 0.18257418583505537f;  // 30^-0.5
          long long off =
              (((long long)(win * NHEAD + head)) * 16 + (intra >> 4)) * 512 +
              frag_idx(intra & 15, d);          // A-frag: row=token, k=d
          qb[off] = (__bf16)c;
        } else if (sel == 1) {
          long long off =
              (((long long)(win * NHEAD + head)) * 16 + (intra >> 4)) * 512 +
              frag_idx(intra & 15, d);          // B-frag: n=key, k=d
          kb[off] = (__bf16)c;
        } else {
          int chunk = intra >> 5, kloc = intra & 31;
          int hdt = d >> 4, nloc = d & 15;
          long long off =
              ((((long long)(win * NHEAD + head)) * 8 + chunk) * 2 + hdt) * 512 +
              frag_idx(nloc, kloc);             // B-frag: n=hd, k=key
          vb[off] = (__bf16)c;
        }
      }
    }
  }
}

// ---------------------------------------------------------------------------
// K2: windowed attention. 1 block/window, 8 waves; wave owns (head, q-tile).
//     Scores via 16 WMMAs (K=32 padded head dim); softmax writes probs into
//     fragment-ordered LDS so attn*V A-operands are contiguous ds loads.
//     K/V fragments are double-buffered across loop iterations.
//     192KB dynamic LDS (WGP budget is 320KB).
// ---------------------------------------------------------------------------
__global__ void __launch_bounds__(256) k_attn(
    const __bf16* __restrict__ qb, const __bf16* __restrict__ kb,
    const __bf16* __restrict__ vb, const float* __restrict__ mask,
    const float* __restrict__ biasT, __bf16* __restrict__ ao) {
  extern __shared__ char smem[];
  const int tid = threadIdx.x, lane = tid & 31, wave = tid >> 5;
  float*  sc = (float*)(smem + wave * 24576);
  __bf16* pr = (__bf16*)(smem + wave * 24576 + 16384);  // [chunk(8)][512] frags
  const int win = blockIdx.x, iw = win & (NWIN - 1);
  const int nl = lane & 15, half = lane >> 4;

  for (int pair = wave; pair < NHEAD * 16; pair += 8) {
    int head = pair / 16, qt = pair % 16;
    const __bf16* hbase_q =
        qb + (((long long)(win * NHEAD + head)) * 16) * 512 + lane * 16;
    const __bf16* hbase_k =
        kb + (((long long)(win * NHEAD + head)) * 16) * 512 + lane * 16;
    const __bf16* hbase_v =
        vb + (((long long)(win * NHEAD + head)) * 16) * 512 + lane * 16;
    v16bf Aq = *(const v16bf*)(hbase_q + qt * 512);

    v16bf Bk = *(const v16bf*)(hbase_k);
    for (int kt = 0; kt < 16; ++kt) {
      v16bf Bcur = Bk;
      if (kt < 15) Bk = *(const v16bf*)(hbase_k + (kt + 1) * 512);
      v8f s = {};
      s = wmma_bf16(Aq, Bcur, s);
#pragma unroll
      for (int j = 0; j < 8; ++j) {
        int m = j + (half << 3);
        int qq = qt * 16 + m, kg = kt * 16 + nl;
        float val = s[j] + biasT[(head << 16) + (qq << 8) + kg] +
                    mask[((long long)iw * NN + qq) * NN + kg];
        sc[m * NN + kg] = val;
      }
    }
    // softmax: lane pair (nl, nl+16) splits row nl into two 128-col halves;
    // normalized probs are written straight into fragment order.
    {
      int r = nl, cb = half * 128;
      float mx = -1e30f;
      for (int c = 0; c < 128; ++c) mx = fmaxf(mx, sc[r * NN + cb + c]);
      mx = fmaxf(mx, __shfl_xor(mx, 16));
      float sum = 0.f;
      for (int c = 0; c < 128; ++c) {
        float e = __expf(sc[r * NN + cb + c] - mx);
        sc[r * NN + cb + c] = e;
        sum += e;
      }
      sum += __shfl_xor(sum, 16);
      float inv = 1.0f / sum;
      for (int c = 0; c < 128; ++c) {
        int cc = cb + c, ch2 = cc >> 5, kloc = cc & 31;
        pr[(ch2 << 9) + frag_idx(r, kloc)] = (__bf16)(sc[r * NN + cc] * inv);
      }
    }
    v8f a0 = {}, a1 = {};
    v16bf B0 = *(const v16bf*)(hbase_v);
    v16bf B1 = *(const v16bf*)(hbase_v + 512);
    for (int ch = 0; ch < 8; ++ch) {
      v16bf C0 = B0, C1 = B1;
      if (ch < 7) {
        B0 = *(const v16bf*)(hbase_v + (ch + 1) * 1024);
        B1 = *(const v16bf*)(hbase_v + (ch + 1) * 1024 + 512);
      }
      v16bf Ap = *(const v16bf*)(pr + (ch << 9) + lane * 16);
      a0 = wmma_bf16(Ap, C0, a0);
      a1 = wmma_bf16(Ap, C1, a1);
    }
#pragma unroll
    for (int j = 0; j < 8; ++j) {
      int m = j + (half << 3);
      int tok = qt * 16 + m;
      long long base = ((long long)win * NN + tok) * DIM + head * HD;
      ao[base + nl] = (__bf16)a0[j];
      if (nl + 16 < HD) ao[base + 16 + nl] = (__bf16)a1[j];
    }
  }
}

// ---------------------------------------------------------------------------
// K3: proj GEMM (180x180) with un-shift gather + residual -> x1 (fp32).
// ---------------------------------------------------------------------------
__global__ void __launch_bounds__(256) k_proj(
    const __bf16* __restrict__ ao, const __bf16* __restrict__ wp,
    const float* __restrict__ pb, const float* __restrict__ x,
    float* __restrict__ x1) {
  __shared__ __align__(32) __bf16 af[6][512];
  const int tid = threadIdx.x;
  const long long g0 = (long long)blockIdx.x * 16;
  for (int idx = tid; idx < 16 * DIM; idx += 256) {
    int r = idx / DIM, c = idx % DIM;
    long long g = g0 + r;
    int b = (int)(g >> 14), rem = (int)(g & 16383);
    int h = rem >> 7, w = rem & 127;
    int hr = (h - SS) & 127, wr = (w - SS) & 127;
    int win = b * NWIN + ((hr >> 4) << 3) + (wr >> 4);
    int intra = ((hr & 15) << 4) + (wr & 15);
    af[c >> 5][frag_idx(r, c & 31)] = ao[((long long)win * NN + intra) * DIM + c];
  }
  for (int idx = tid; idx < 16 * 12; idx += 256) {
    int r = idx / 12, k = DIM + idx % 12;
    af[5][frag_idx(r, k & 31)] = (__bf16)0.0f;
  }
  __syncthreads();
  const int lane = tid & 31, wave = tid >> 5;
  v16bf A[6];
#pragma unroll
  for (int kk = 0; kk < 6; ++kk) A[kk] = *(const v16bf*)&af[kk][lane * 16];
  const int nl = lane & 15, half = lane >> 4;
  for (int nt = wave; nt < 12; nt += 8) {
    int n = nt * 16 + nl;
    const __bf16* bp = wp + (long long)nt * 6 * 512 + lane * 16;
    v16bf Bf[6];
#pragma unroll
    for (int kk = 0; kk < 6; ++kk) Bf[kk] = *(const v16bf*)(bp + (kk << 9));
    v8f acc = {};
#pragma unroll
    for (int kk = 0; kk < 6; ++kk) acc = wmma_bf16(A[kk], Bf[kk], acc);
    if (n < DIM) {
      float bias = pb[n];
#pragma unroll
      for (int j = 0; j < 8; ++j) {
        long long g = g0 + j + (half << 3);
        x1[g * DIM + n] = acc[j] + bias + x[g * DIM + n];
      }
    }
  }
}

// ---------------------------------------------------------------------------
// K4: LayerNorm2 + fc1 GEMM (180(pad192) x 360) + exact GELU -> y (bf16).
// ---------------------------------------------------------------------------
__global__ void __launch_bounds__(256) k_ln_fc1(
    const float* __restrict__ x1, const float* __restrict__ nw2,
    const float* __restrict__ nb2, const __bf16* __restrict__ w1,
    const float* __restrict__ b1, __bf16* __restrict__ y) {
  __shared__ float xs[16][DIM];
  __shared__ __align__(32) __bf16 af[6][512];
  const int tid = threadIdx.x;
  const long long g0 = (long long)blockIdx.x * 16;
  for (int idx = tid; idx < 16 * DIM; idx += 256) {
    int r = idx / DIM, c = idx % DIM;
    xs[r][c] = x1[(g0 + r) * DIM + c];
  }
  __syncthreads();
  if (tid < 16) {
    int r = tid;
    float m = 0.f;
    for (int c = 0; c < DIM; ++c) m += xs[r][c];
    m *= (1.0f / DIM);
    float v = 0.f;
    for (int c = 0; c < DIM; ++c) { float d = xs[r][c] - m; v += d * d; }
    float rs = rsqrtf(v * (1.0f / DIM) + 1e-5f);
    for (int k = 0; k < 192; ++k) {
      float val = (k < DIM) ? (xs[r][k] - m) * rs * nw2[k] + nb2[k] : 0.f;
      af[k >> 5][frag_idx(r, k & 31)] = (__bf16)val;
    }
  }
  __syncthreads();
  const int lane = tid & 31, wave = tid >> 5;
  v16bf A[6];
#pragma unroll
  for (int kk = 0; kk < 6; ++kk) A[kk] = *(const v16bf*)&af[kk][lane * 16];
  const int nl = lane & 15, half = lane >> 4;
  for (int nt = wave; nt < 23; nt += 8) {
    int n = nt * 16 + nl;
    const __bf16* bp = w1 + (long long)nt * 6 * 512 + lane * 16;
    v16bf Bf[6];
#pragma unroll
    for (int kk = 0; kk < 6; ++kk) Bf[kk] = *(const v16bf*)(bp + (kk << 9));
    v8f acc = {};
#pragma unroll
    for (int kk = 0; kk < 6; ++kk) acc = wmma_bf16(A[kk], Bf[kk], acc);
    if (n < HID) {
      float bias = b1[n];
#pragma unroll
      for (int j = 0; j < 8; ++j) {
        long long g = g0 + j + (half << 3);
        y[g * HID + n] = (__bf16)gelu_f(acc[j] + bias);
      }
    }
  }
}

// ---------------------------------------------------------------------------
// K5: depthwise 5x5 conv + GELU + add -> z (bf16). Channel-contiguous loads.
// ---------------------------------------------------------------------------
__global__ void __launch_bounds__(256) k_dwconv(
    const __bf16* __restrict__ y, const float* __restrict__ dww,
    const float* __restrict__ dwb, __bf16* __restrict__ z) {
  long long idx = (long long)blockIdx.x * 256 + threadIdx.x;
  int ch = (int)(idx % HID);
  long long t = idx / HID;
  int w = (int)(t & 127), h = (int)((t >> 7) & 127), b = (int)(t >> 14);
  float acc = dwb[ch];
#pragma unroll
  for (int dh = -2; dh <= 2; ++dh) {
    int hh = h + dh;
    if (hh < 0 || hh >= 128) continue;
#pragma unroll
    for (int dw = -2; dw <= 2; ++dw) {
      int ww2 = w + dw;
      if (ww2 < 0 || ww2 >= 128) continue;
      long long src = ((((long long)b << 7) + hh) << 7) + ww2;
      acc += (float)y[src * HID + ch] * dww[ch * 25 + (dh + 2) * 5 + (dw + 2)];
    }
  }
  z[idx] = (__bf16)((float)y[idx] + gelu_f(acc));
}

// ---------------------------------------------------------------------------
// K6: fc2 GEMM (360(pad384) x 180) + residual -> out (fp32).
// ---------------------------------------------------------------------------
__global__ void __launch_bounds__(256) k_fc2(
    const __bf16* __restrict__ z, const __bf16* __restrict__ w2,
    const float* __restrict__ b2, const float* __restrict__ x1,
    float* __restrict__ out) {
  __shared__ __align__(32) __bf16 af[12][512];
  const int tid = threadIdx.x;
  const long long g0 = (long long)blockIdx.x * 16;
  for (int idx = tid; idx < 16 * HID; idx += 256) {
    int r = idx / HID, c = idx % HID;
    af[c >> 5][frag_idx(r, c & 31)] = z[(g0 + r) * HID + c];
  }
  for (int idx = tid; idx < 16 * 24; idx += 256) {
    int r = idx / 24, k = HID + idx % 24;
    af[11][frag_idx(r, k & 31)] = (__bf16)0.0f;
  }
  __syncthreads();
  const int lane = tid & 31, wave = tid >> 5;
  v16bf A[12];
#pragma unroll
  for (int kk = 0; kk < 12; ++kk) A[kk] = *(const v16bf*)&af[kk][lane * 16];
  const int nl = lane & 15, half = lane >> 4;
  for (int nt = wave; nt < 12; nt += 8) {
    int n = nt * 16 + nl;
    const __bf16* bp = w2 + (long long)nt * 12 * 512 + lane * 16;
    // Two preload groups of 6 to bound live B registers (~48 VGPRs).
    v8f acc = {};
#pragma unroll
    for (int half_k = 0; half_k < 2; ++half_k) {
      v16bf Bf[6];
#pragma unroll
      for (int kk = 0; kk < 6; ++kk)
        Bf[kk] = *(const v16bf*)(bp + ((half_k * 6 + kk) << 9));
#pragma unroll
      for (int kk = 0; kk < 6; ++kk)
        acc = wmma_bf16(A[half_k * 6 + kk], Bf[kk], acc);
    }
    if (n < DIM) {
      float bias = b2[n];
#pragma unroll
      for (int j = 0; j < 8; ++j) {
        long long g = g0 + j + (half << 3);
        out[g * DIM + n] = acc[j] + bias + x1[g * DIM + n];
      }
    }
  }
}

// ---------------------------------------------------------------------------
// Workspace layout (bytes), reuse exploits stream ordering:
//   [0,            50331648)  q fragments (bf16)   | later reused: x1 fp32
//   [50331648,    100663296)  k fragments (bf16)   |   (94371840 B)
//   [100663296,   150994944)  v fragments (bf16)
//   [150994944,   198180864)  attention out (bf16) | later reused by y
//   [94371840,    188743680)  y = gelu(fc1) (bf16)
//   [188743680,   283115520)  z = y + gelu(dwconv) (bf16)
//   [283115520, ...)          packed weights + bias table (~2.1 MB)
// Peak ws requirement: ~285 MB.
// ---------------------------------------------------------------------------
extern "C" void kernel_launch(void* const* d_in, const int* in_sizes, int n_in,
                              void* d_out, int out_size, void* d_ws,
                              size_t ws_size, hipStream_t stream) {
  (void)in_sizes; (void)n_in; (void)out_size; (void)ws_size;
  const float* x    = (const float*)d_in[0];
  const float* amsk = (const float*)d_in[1];
  const int*   rpi  = (const int*)d_in[2];
  const float* n1w  = (const float*)d_in[3];
  const float* n1b  = (const float*)d_in[4];
  const float* wqkv = (const float*)d_in[5];
  const float* bqkv = (const float*)d_in[6];
  const float* rpb  = (const float*)d_in[7];
  const float* pw   = (const float*)d_in[8];
  const float* pb   = (const float*)d_in[9];
  const float* n2w  = (const float*)d_in[10];
  const float* n2b  = (const float*)d_in[11];
  const float* w1   = (const float*)d_in[12];
  const float* b1   = (const float*)d_in[13];
  const float* dww  = (const float*)d_in[14];
  const float* dwb  = (const float*)d_in[15];
  const float* w2   = (const float*)d_in[16];
  const float* b2   = (const float*)d_in[17];

  char* ws = (char*)d_ws;
  __bf16* qb = (__bf16*)(ws + 0LL);
  __bf16* kb = (__bf16*)(ws + 50331648LL);
  __bf16* vb = (__bf16*)(ws + 100663296LL);
  __bf16* ao = (__bf16*)(ws + 150994944LL);
  float*  x1 = (float*)(ws + 0LL);          // reuses q/k region after K2
  __bf16* y  = (__bf16*)(ws + 94371840LL);  // reuses v/ao region after K3
  __bf16* z  = (__bf16*)(ws + 188743680LL);
  __bf16* wq_p = (__bf16*)(ws + 283115520LL);  // 34*6*512*2  = 208896 B
  __bf16* wp_p = (__bf16*)(ws + 283324416LL);  // 12*6*512*2  =  73728 B
  __bf16* w1_p = (__bf16*)(ws + 283398144LL);  // 23*6*512*2  = 141312 B
  __bf16* w2_p = (__bf16*)(ws + 283539456LL);  // 12*12*512*2 = 147456 B
  float*  bT   = (float*)(ws + 283686912LL);   // 6*256*256*4 = 1572864 B
  float* out = (float*)d_out;

  // Deterministic zero for head-dim pad lanes (d=30,31) inside q/k/v frags.
  hipMemsetAsync(ws, 0, 150994944LL, stream);

  // Weight packing + bias table (L2-resident, reused by all GEMM blocks).
  k_pack_w<<<(34 * 6 * 512 + 255) / 256, 256, 0, stream>>>(wqkv, wq_p, DIM,
                                                           QKVN, 34, 6);
  k_pack_w<<<(12 * 6 * 512 + 255) / 256, 256, 0, stream>>>(pw, wp_p, DIM, DIM,
                                                           12, 6);
  k_pack_w<<<(23 * 6 * 512 + 255) / 256, 256, 0, stream>>>(w1, w1_p, DIM, HID,
                                                           23, 6);
  k_pack_w<<<(12 * 12 * 512 + 255) / 256, 256, 0, stream>>>(w2, w2_p, HID, DIM,
                                                            12, 12);
  k_bias_tab<<<(NHEAD * NN * NN + 255) / 256, 256, 0, stream>>>(rpi, rpb, bT);

  k_ln_qkv<<<8192, 256, 0, stream>>>(x, n1w, n1b, wq_p, bqkv, qb, kb, vb);
  k_attn<<<512, 256, 196608, stream>>>(qb, kb, vb, amsk, bT, ao);
  k_proj<<<8192, 256, 0, stream>>>(ao, wp_p, pb, x, x1);
  k_ln_fc1<<<8192, 256, 0, stream>>>(x1, n2w, n2b, w1_p, b1, y);
  k_dwconv<<<184320, 256, 0, stream>>>(y, dww, dwb, z);
  k_fc2<<<8192, 256, 0, stream>>>(z, w2_p, b2, x1, out);
}
